// Pinnacle_69999376990620
// MI455X (gfx1250) — compile-verified, matrix operand order checked by reading, not run on hardware
//
#include <hip/hip_runtime.h>

#define CC 8
#define NN 20000
#define EE 200000
#define PP 3
#define MM 64
#define EMM 512
#define PMM 2
#define DD 128
#define HH 8
#define HIDD 16
#define SEMM 8
#define NEGS 0.2f

typedef __attribute__((ext_vector_type(16))) __bf16 v16bf;
typedef __attribute__((ext_vector_type(8)))  float  v8f;
typedef __attribute__((ext_vector_type(4)))  float  f4;
typedef __attribute__((ext_vector_type(4)))  int    v4i;

#if __has_builtin(__builtin_amdgcn_global_load_async_to_lds_b128)
#define HAVE_ASYNC_LDS 1
#else
#define HAVE_ASYNC_LDS 0
#endif

static inline int ceil_div(int a, int b) { return (a + b - 1) / b; }

// ---------- device helpers ----------

__device__ inline float lrelu02(float x) { return x > 0.f ? x : NEGS * x; }

__device__ inline void atomicMaxF(float* addr, float v) {
  unsigned int* u = (unsigned int*)addr;
  unsigned int old = *u;
  while (__uint_as_float(old) < v) {
    unsigned int assumed = old;
    old = atomicCAS(u, assumed, __float_as_uint(v));
    if (old == assumed) break;
  }
}

#if HAVE_ASYNC_LDS
// Stage this wave's 4KB of pre-swizzled B fragments into LDS via the gfx1250
// async-to-LDS path (tracked by ASYNCcnt). Each lane issues 8 x b128 transfers.
__device__ inline void stage_B_async(const __bf16* Bp, __bf16* Blds, int wave, int lane) {
  const char* g = (const char*)Bp + (size_t)wave * 4096;
  char* l = (char*)Blds + (size_t)wave * 4096;
#pragma unroll
  for (int j = 0; j < 8; ++j) {
    int off = (j * 32 + lane) * 16;
    __builtin_amdgcn_global_load_async_to_lds_b128(
        (__attribute__((address_space(1))) v4i*)(g + off),
        (__attribute__((address_space(3))) v4i*)(l + off), 0, 0);
  }
#if __has_builtin(__builtin_amdgcn_s_wait_asynccnt)
  __builtin_amdgcn_s_wait_asynccnt(0);
#else
  asm volatile("s_wait_asynccnt 0x0" ::: "memory");
#endif
  asm volatile("" ::: "memory");  // keep LDS reads below the wait
}
#endif

// ---------- generic fills ----------

__global__ void fill_kernel(float* p, float v, int n) {
  int i = blockIdx.x * blockDim.x + threadIdx.x;
  if (i < n) p[i] = v;
}

// ---------- weight pre-swizzle into WMMA B-fragment order ----------
// For each matrix: Bp[((tile*4 + kstep)*32 + lane)*16 + i] holds the bf16 B-fragment
// element i for that (N-tile, K-step, lane). One thread per (tile,kstep,lane) = 1024.
__global__ void pack_B_kernel(const float* __restrict__ W, __bf16* __restrict__ Bp) {
  int t = blockIdx.x * blockDim.x + threadIdx.x;
  if (t >= 1024) return;
  int lane = t & 31, ks = (t >> 5) & 3, tile = t >> 7;
  int hf = lane >> 4, idx = lane & 15;
  int col = tile * 16 + idx;
  int k0 = ks * 32;
  __bf16* o = Bp + (size_t)t * 16;
#pragma unroll
  for (int i = 0; i < 8; ++i) {
    o[i]     = (__bf16)W[(size_t)(k0 + 8 * hf + i) * DD + col];
    o[8 + i] = (__bf16)W[(size_t)(k0 + 16 + 8 * hf + i) * DD + col];
  }
}

// ---------- WMMA GEMM: H = X(f32)[n,128] @ W(pre-swizzled bf16), fused head dots ----------
// block = 256 threads = 8 waves; wave w computes the 16x16 tile covering head w.
// n must be a multiple of 16 (true for 20000 and 64) -> no bounds checks.
// A fragment per lane: rows = lane%16; K runs [k0+8*hf, +8) and [k0+16+8*hf, +8).

__global__ void gemm_h_kernel(const float* __restrict__ X,
                              const __bf16* __restrict__ Bp,
                              float* __restrict__ Hout,
                              const float* __restrict__ as,
                              const float* __restrict__ ad,
                              float* __restrict__ Ss,
                              float* __restrict__ Sd,
                              int n) {
  const int lane = threadIdx.x & 31;
  const int wave = threadIdx.x >> 5;   // head / N-tile id (0..7)
  const int hf   = lane >> 4;
  const int idx  = lane & 15;
  const int m0   = blockIdx.x * 16;
  const int row  = m0 + idx;
  const int col  = wave * 16 + idx;

#if HAVE_ASYNC_LDS
  __shared__ __bf16 Blds[8 * 2048];   // 32 KB: 4 KB of B fragments per wave
  stage_B_async(Bp, Blds, wave, lane);
#endif

  const f4* __restrict__ Xr = (const f4*)(X + (size_t)row * DD);
  // prefetch next block's row (gfx1250 global_prefetch path)
  __builtin_prefetch(X + ((size_t)row + 16) * DD, 0, 1);

  v8f acc = {};
#pragma unroll
  for (int ks = 0; ks < 4; ++ks) {
    const int b0 = (ks * 32 + 8 * hf) >> 2;        // f4 index of run 0
    const f4 x0 = Xr[b0], x1 = Xr[b0 + 1];
    const f4 x2 = Xr[b0 + 4], x3 = Xr[b0 + 5];     // run 1 = +16 floats
    v16bf a;
#pragma unroll
    for (int j = 0; j < 4; ++j) {
      a[j]      = (__bf16)x0[j];
      a[4 + j]  = (__bf16)x1[j];
      a[8 + j]  = (__bf16)x2[j];
      a[12 + j] = (__bf16)x3[j];
    }
#if HAVE_ASYNC_LDS
    const v16bf b = *(const v16bf*)(Blds + (size_t)((wave * 4 + ks) * 32 + lane) * 16);
#else
    const v16bf b = *(const v16bf*)(Bp + (size_t)((wave * 4 + ks) * 32 + lane) * 16);
#endif
    acc = __builtin_amdgcn_wmma_f32_16x16x32_bf16(false, a, false, b, (short)0, acc,
                                                  false, false);
  }
#pragma unroll
  for (int r = 0; r < 8; ++r)
    Hout[(size_t)(m0 + r + 8 * hf) * DD + col] = acc[r];

  // fused per-head attention dots: s_src[m][head] = sum_c h[m][head*16+c]*a_src[head][c]
  const float asv = as[wave * HIDD + idx];
  const float adv = ad[wave * HIDD + idx];
#pragma unroll
  for (int r = 0; r < 8; ++r) {
    float vs = acc[r] * asv;
    float vd = acc[r] * adv;
#pragma unroll
    for (int off = 1; off < 16; off <<= 1) {  // reduce within the 16-lane half
      vs += __shfl_xor(vs, off, 32);
      vd += __shfl_xor(vd, off, 32);
    }
    if (idx == 0) {
      const int m = m0 + r + 8 * hf;
      Ss[(size_t)m * HH + wave] = vs;
      Sd[(size_t)m * HH + wave] = vd;
    }
  }
}

// plain WMMA GEMM (no epilogue): O = X @ W (pre-swizzled bf16)
__global__ void gemm_plain_kernel(const float* __restrict__ X,
                                  const __bf16* __restrict__ Bp,
                                  float* __restrict__ O, int n) {
  const int lane = threadIdx.x & 31;
  const int wave = threadIdx.x >> 5;
  const int hf   = lane >> 4;
  const int idx  = lane & 15;
  const int m0   = blockIdx.x * 16;
  const int row  = m0 + idx;
  const int col  = wave * 16 + idx;

#if HAVE_ASYNC_LDS
  __shared__ __bf16 Blds[8 * 2048];
  stage_B_async(Bp, Blds, wave, lane);
#endif

  const f4* __restrict__ Xr = (const f4*)(X + (size_t)row * DD);

  v8f acc = {};
#pragma unroll
  for (int ks = 0; ks < 4; ++ks) {
    const int b0 = (ks * 32 + 8 * hf) >> 2;
    const f4 x0 = Xr[b0], x1 = Xr[b0 + 1];
    const f4 x2 = Xr[b0 + 4], x3 = Xr[b0 + 5];
    v16bf a;
#pragma unroll
    for (int j = 0; j < 4; ++j) {
      a[j]      = (__bf16)x0[j];
      a[4 + j]  = (__bf16)x1[j];
      a[8 + j]  = (__bf16)x2[j];
      a[12 + j] = (__bf16)x3[j];
    }
#if HAVE_ASYNC_LDS
    const v16bf b = *(const v16bf*)(Blds + (size_t)((wave * 4 + ks) * 32 + lane) * 16);
#else
    const v16bf b = *(const v16bf*)(Bp + (size_t)((wave * 4 + ks) * 32 + lane) * 16);
#endif
    acc = __builtin_amdgcn_wmma_f32_16x16x32_bf16(false, a, false, b, (short)0, acc,
                                                  false, false);
  }
#pragma unroll
  for (int r = 0; r < 8; ++r)
    O[(size_t)(m0 + r + 8 * hf) * DD + col] = acc[r];
}

// ---------- edge-wise attention kernels (thread per (edge, head)) ----------

__global__ void edge_logit_max_kernel(const int* __restrict__ src, const int* __restrict__ dst,
                                      const float* __restrict__ Ss, const float* __restrict__ Sd,
                                      float* __restrict__ mx, int e) {
  int t = blockIdx.x * blockDim.x + threadIdx.x;
  if (t >= e * HH) return;
  int eid = t >> 3, h = t & 7;
  int s = src[eid], d = dst[eid];
  float x = lrelu02(Ss[(size_t)s * HH + h] + Sd[(size_t)d * HH + h]);
  atomicMaxF(&mx[(size_t)d * HH + h], x);
}

__global__ void edge_den_kernel(const int* __restrict__ src, const int* __restrict__ dst,
                                const float* __restrict__ Ss, const float* __restrict__ Sd,
                                const float* __restrict__ mx, float* __restrict__ den, int e) {
  int t = blockIdx.x * blockDim.x + threadIdx.x;
  if (t >= e * HH) return;
  int eid = t >> 3, h = t & 7;
  int s = src[eid], d = dst[eid];
  float x = lrelu02(Ss[(size_t)s * HH + h] + Sd[(size_t)d * HH + h]);
  float p = __expf(x - mx[(size_t)d * HH + h]);
  atomicAdd(&den[(size_t)d * HH + h], p);
}

__global__ void edge_scatter_kernel(const int* __restrict__ src, const int* __restrict__ dst,
                                    const float* __restrict__ Ss, const float* __restrict__ Sd,
                                    const float* __restrict__ mx, const float* __restrict__ den,
                                    const float* __restrict__ Hfeat, float* __restrict__ out,
                                    int e) {
  int t = blockIdx.x * blockDim.x + threadIdx.x;
  if (t >= e * HH) return;
  int eid = t >> 3, h = t & 7;
  int s = src[eid], d = dst[eid];
  float x = lrelu02(Ss[(size_t)s * HH + h] + Sd[(size_t)d * HH + h]);
  float p = __expf(x - mx[(size_t)d * HH + h]);
  float alpha = p / (den[(size_t)d * HH + h] + 1e-16f);
  const float* hs = Hfeat + (size_t)s * DD + h * HIDD;
  float* o = out + (size_t)d * DD + h * HIDD;
#pragma unroll
  for (int k = 0; k < HIDD; ++k) atomicAdd(&o[k], hs[k] * alpha);
}

// ---------- semantic attention ----------

__global__ void sem_score_kernel(const float* __restrict__ z, const float* __restrict__ sW,
                                 const float* __restrict__ sb, const float* __restrict__ sq,
                                 float* __restrict__ score, int n, int P) {
  int t = blockIdx.x * blockDim.x + threadIdx.x;
  if (t >= n * P) return;
  int p = t / n, i = t - p * n;
  const float* zr = z + ((size_t)p * n + i) * DD;
  float acc = 0.f;
  for (int s2 = 0; s2 < SEMM; ++s2) {
    float d = sb[s2];
    for (int k = 0; k < DD; ++k) d += zr[k] * sW[k * SEMM + s2];
    acc += tanhf(d) * sq[s2];
  }
  atomicAdd(&score[p], acc);
}

__global__ void softmax_w_kernel(const float* __restrict__ score, float* __restrict__ w,
                                 int P, float invn) {
  if (threadIdx.x != 0 || blockIdx.x != 0) return;
  float m = -1e30f;
  for (int p = 0; p < P; ++p) {
    float s = score[p] * invn;
    m = s > m ? s : m;
  }
  float sum = 0.f;
  for (int p = 0; p < P; ++p) {
    float e2 = __expf(score[p] * invn - m);
    w[p] = e2;
    sum += e2;
  }
  for (int p = 0; p < P; ++p) w[p] /= sum;
}

__global__ void sem_combine_kernel(const float* __restrict__ z, const float* __restrict__ w,
                                   float* __restrict__ out, int n, int P) {
  size_t t = (size_t)blockIdx.x * blockDim.x + threadIdx.x;
  if (t >= (size_t)n * DD) return;
  float acc = 0.f;
  for (int p = 0; p < P; ++p) acc += w[p] * z[(size_t)p * n * DD + t];
  out[t] = acc;
}

// ---------- misc elementwise ----------

__global__ void add_pc_lrelu_kernel(float* __restrict__ h, const float* __restrict__ pc,
                                    const int* __restrict__ tn, int c, int n) {
  size_t t = (size_t)blockIdx.x * blockDim.x + threadIdx.x;
  if (t >= (size_t)n * DD) return;
  int col = (int)(t & (DD - 1));
  float v = h[t] + pc[(size_t)tn[c] * DD + col];
  h[t] = lrelu02(v);
}

__global__ void ln_lrelu_kernel(float* __restrict__ x, const float* __restrict__ g,
                                const float* __restrict__ b, int n) {
  int i = blockIdx.x * blockDim.x + threadIdx.x;
  if (i >= n) return;
  float* r = x + (size_t)i * DD;
  float mu = 0.f;
  for (int k = 0; k < DD; ++k) mu += r[k];
  mu *= (1.f / DD);
  float v = 0.f;
  for (int k = 0; k < DD; ++k) {
    float d2 = r[k] - mu;
    v += d2 * d2;
  }
  v *= (1.f / DD);
  float is = rsqrtf(v + 1e-5f);
  for (int k = 0; k < DD; ++k) {
    float y = g[k] * (r[k] - mu) * is + b[k];
    r[k] = y > 0.f ? y : 0.01f * y;
  }
}

__global__ void bn_stats_kernel(const float* __restrict__ x, float* __restrict__ sums,
                                float* __restrict__ sumsq, int n) {
  size_t t = (size_t)blockIdx.x * blockDim.x + threadIdx.x;
  if (t >= (size_t)n * DD) return;
  int k = (int)(t & (DD - 1));
  float v = x[t];
  atomicAdd(&sums[k], v);
  atomicAdd(&sumsq[k], v * v);
}

__global__ void bn_apply_kernel(float* __restrict__ x, const float* __restrict__ sums,
                                const float* __restrict__ sumsq, const float* __restrict__ g,
                                const float* __restrict__ b, int n) {
  size_t t = (size_t)blockIdx.x * blockDim.x + threadIdx.x;
  if (t >= (size_t)n * DD) return;
  int k = (int)(t & (DD - 1));
  float mu = sums[k] / n;
  float var = sumsq[k] / n - mu * mu;
  x[t] = g[k] * (x[t] - mu) * rsqrtf(var + 1e-5f) + b[k];
}

// ---------- host orchestration helpers ----------

static void run_gat(const float* X, int n, const int* src, const int* dst, int e,
                    const __bf16* Bp, const float* as, const float* ad, float* out,
                    bool zero_out, float* Htmp, float* Ss, float* Sd, float* mxb, float* den,
                    hipStream_t st) {
  gemm_h_kernel<<<n / 16, 256, 0, st>>>(X, Bp, Htmp, as, ad, Ss, Sd, n);
  fill_kernel<<<ceil_div(n * HH, 256), 256, 0, st>>>(mxb, -1e30f, n * HH);
  fill_kernel<<<ceil_div(n * HH, 256), 256, 0, st>>>(den, 0.f, n * HH);
  if (zero_out) fill_kernel<<<ceil_div(n * DD, 256), 256, 0, st>>>(out, 0.f, n * DD);
  int t = e * HH;
  edge_logit_max_kernel<<<ceil_div(t, 256), 256, 0, st>>>(src, dst, Ss, Sd, mxb, e);
  edge_den_kernel<<<ceil_div(t, 256), 256, 0, st>>>(src, dst, Ss, Sd, mxb, den, e);
  edge_scatter_kernel<<<ceil_div(t, 256), 256, 0, st>>>(src, dst, Ss, Sd, mxb, den, Htmp, out, e);
}

static void run_han(const float* X, int n, const int* mp_base, int P, int e, const __bf16* Bp,
                    const float* as, const float* ad, const float* sW, const float* sb,
                    const float* sq, float* z, float* out, float* score, float* Htmp, float* Ss,
                    float* Sd, float* mxb, float* den, hipStream_t st) {
  for (int p = 0; p < P; ++p) {
    const int* s = mp_base + (size_t)p * 2 * e;
    run_gat(X, n, s, s + e, e, Bp, as, ad, z + (size_t)p * n * DD, true, Htmp, Ss, Sd, mxb, den,
            st);
  }
  fill_kernel<<<1, 32, 0, st>>>(score, 0.f, 8);
  sem_score_kernel<<<ceil_div(P * n, 128), 128, 0, st>>>(z, sW, sb, sq, score, n, P);
  softmax_w_kernel<<<1, 1, 0, st>>>(score, score + 8, P, 1.0f / n);
  sem_combine_kernel<<<ceil_div(n * DD, 256), 256, 0, st>>>(z, score + 8, out, n, P);
}

// ---------- entry ----------

extern "C" void kernel_launch(void* const* d_in, const int* in_sizes, int n_in, void* d_out,
                              int out_size, void* d_ws, size_t ws_size, hipStream_t stream) {
  const float* ppi_x  = (const float*)d_in[0];
  const float* mg_x   = (const float*)d_in[1];
  const int*   ppi_mp = (const int*)d_in[2];
  const int*   mg_mp  = (const int*)d_in[3];
  const int*   ppi_ei = (const int*)d_in[4];
  const int*   tn     = (const int*)d_in[5];
  const float* W_gat  = (const float*)d_in[6];
  const float* a_src  = (const float*)d_in[7];
  const float* a_dst  = (const float*)d_in[8];
  const float* sem_W  = (const float*)d_in[9];
  const float* sem_b  = (const float*)d_in[10];
  const float* sem_q  = (const float*)d_in[11];
  const float* W_pc   = (const float*)d_in[12];
  const float* ln_g   = (const float*)d_in[13];
  const float* ln_b   = (const float*)d_in[14];
  const float* bn_g   = (const float*)d_in[15];
  const float* bn_b   = (const float*)d_in[16];
  float* out = (float*)d_out;
  (void)in_sizes; (void)n_in; (void)out_size; (void)ws_size;

  char* wp = (char*)d_ws;
  auto alloc = [&](size_t bytes) -> void* {
    void* p = (void*)wp;
    wp += (bytes + 255) & ~(size_t)255;
    return p;
  };
  float* pxA   = (float*)alloc((size_t)CC * NN * DD * 4);
  float* pxB   = (float*)alloc((size_t)CC * NN * DD * 4);
  float* mgA   = (float*)alloc((size_t)MM * DD * 4);
  float* Htmp  = (float*)alloc((size_t)NN * DD * 4);
  float* z     = (float*)alloc((size_t)PP * NN * DD * 4);
  float* Ss    = (float*)alloc((size_t)NN * HH * 4);
  float* Sd    = (float*)alloc((size_t)NN * HH * 4);
  float* mxb   = (float*)alloc((size_t)NN * HH * 4);
  float* den   = (float*)alloc((size_t)NN * HH * 4);
  float* pc    = (float*)alloc((size_t)MM * DD * 4);
  float* score = (float*)alloc(64);
  float* bnsum = (float*)alloc(DD * 4);
  float* bnsq  = (float*)alloc(DD * 4);
  __bf16* Wb   = (__bf16*)alloc((size_t)8 * DD * DD * 2);  // 8 pre-swizzled matrices

  // pre-swizzle all GEMM weights into WMMA B-fragment layout (6 GAT mats + 2 W_pc mats)
  for (int i = 0; i < 6; ++i)
    pack_B_kernel<<<4, 256, 0, stream>>>(W_gat + (size_t)i * DD * DD,
                                         Wb + (size_t)i * DD * DD);
  for (int i = 0; i < 2; ++i)
    pack_B_kernel<<<4, 256, 0, stream>>>(W_pc + (size_t)i * DD * DD,
                                         Wb + (size_t)(6 + i) * DD * DD);

  auto Wset = [&](int i) { return Wb + (size_t)i * DD * DD; };
  auto As = [&](int i) { return a_src + (size_t)i * HH * HIDD; };
  auto Ad = [&](int i) { return a_dst + (size_t)i * HH * HIDD; };
  auto SW = [&](int i) { return sem_W + (size_t)i * DD * SEMM; };
  auto SB = [&](int i) { return sem_b + (size_t)i * SEMM; };
  auto SQ = [&](int i) { return sem_q + (size_t)i * SEMM; };

  float* out_mg = out + (size_t)CC * NN * DD;

  auto conv_up = [&](const float* pxin, const float* mgin, int i_ppi, int i_mg,
                     const __bf16* wpc, float* pxout, float* mgout) {
    // metagraph HAN
    run_han(mgin, MM, mg_mp, PMM, EMM, Wset(i_mg), As(i_mg), Ad(i_mg), SW(i_mg), SB(i_mg),
            SQ(i_mg), z, mgout, score, Htmp, Ss, Sd, mxb, den, stream);
    // pc = mgz @ wpc
    gemm_plain_kernel<<<MM / 16, 256, 0, stream>>>(mgout, wpc, pc, MM);
    for (int c = 0; c < CC; ++c) {
      const float* xin = pxin + (size_t)c * NN * DD;
      float* xo = pxout + (size_t)c * NN * DD;
      const int* mp = ppi_mp + (size_t)c * PP * 2 * EE;
      run_han(xin, NN, mp, PP, EE, Wset(i_ppi), As(i_ppi), Ad(i_ppi), SW(i_ppi), SB(i_ppi),
              SQ(i_ppi), z, xo, score, Htmp, Ss, Sd, mxb, den, stream);
      const int* ei = ppi_ei + (size_t)c * 2 * EE;
      run_gat(xin, NN, ei, ei + EE, EE, Wset(i_ppi), As(i_ppi), Ad(i_ppi), xo,
              /*zero_out=*/false, Htmp, Ss, Sd, mxb, den, stream);  // accumulate into HAN out
      add_pc_lrelu_kernel<<<ceil_div(NN * DD, 256), 256, 0, stream>>>(xo, pc, tn, c, NN);
    }
  };

  auto conv_down = [&](const float* pxin, int i, float* pxout) {
    for (int c = 0; c < CC; ++c) {
      run_han(pxin + (size_t)c * NN * DD, NN, ppi_mp + (size_t)c * PP * 2 * EE, PP, EE, Wset(i),
              As(i), Ad(i), SW(i), SB(i), SQ(i), z, pxout + (size_t)c * NN * DD, score, Htmp, Ss,
              Sd, mxb, den, stream);
    }
  };

  auto norm_chain = [&](float* x, int n) {
    ln_lrelu_kernel<<<ceil_div(n, 128), 128, 0, stream>>>(x, ln_g, ln_b, n);
    fill_kernel<<<1, 128, 0, stream>>>(bnsum, 0.f, DD);
    fill_kernel<<<1, 128, 0, stream>>>(bnsq, 0.f, DD);
    bn_stats_kernel<<<ceil_div(n * DD, 256), 256, 0, stream>>>(x, bnsum, bnsq, n);
    bn_apply_kernel<<<ceil_div(n * DD, 256), 256, 0, stream>>>(x, bnsum, bnsq, bn_g, bn_b, n);
  };

  // ---- layer 1 ----
  conv_up(ppi_x, mg_x, 0, 1, Wset(6), pxA, mgA);
  conv_down(pxA, 2, pxB);
  for (int c = 0; c < CC; ++c) norm_chain(pxB + (size_t)c * NN * DD, NN);
  norm_chain(mgA, MM);

  // ---- layer 2 ----
  conv_up(pxB, mgA, 3, 4, Wset(7), pxA, out_mg);  // mgz of layer 2 is the final mx output
  conv_down(pxA, 5, out);                         // final px written straight into d_out
}
